// MambaBlock_1082331759237
// MI455X (gfx1250) — compile-verified, matrix operand order checked by reading, not run on hardware
//
#include <hip/hip_runtime.h>
#include <math.h>

#define SEQ    4096
#define DM     256
#define DI     512
#define DS     16
#define DTR    16
#define NBATCH 8
#define BL     (NBATCH * SEQ)   /* 32768 rows */
#define NCH    8                /* scan chunks per sequence */
#define CHL    (SEQ / NCH)      /* 512 steps per chunk */
#define EPSF   1e-5f

typedef __attribute__((ext_vector_type(16))) _Float16 v16h;
typedef __attribute__((ext_vector_type(8)))  _Float16 v8h;
typedef __attribute__((ext_vector_type(8)))  float    v8f;

__device__ __forceinline__ float sigmf(float x) { return 1.0f / (1.0f + __expf(-x)); }

// ---------------------------------------------------------------------------
// Pack a row-major fp32 weight [K][N] into f16 WMMA B-fragments.
// Fragment (kt, nt) is 512 halves; lane l owns 16 contiguous halves at l*16.
// Element e of lane l maps to k = kt*32 + (e&7) + 8*(l>>4) + 16*(e>>3),
//                           n = nt*16 + (l&15)   (CDNA5 16-bit B layout).
// ---------------------------------------------------------------------------
__global__ __launch_bounds__(256) void pack_b_kernel(
    const float* __restrict__ B, _Float16* __restrict__ Bp, int K, int N, int Np) {
  int g = blockIdx.x * 256 + threadIdx.x;
  int total = (K >> 5) * (Np >> 4) * 512;
  if (g >= total) return;
  int frag = g >> 9, within = g & 511;
  int lane = within >> 4, e = within & 15;
  int nfrag = Np >> 4;
  int kt = frag / nfrag, nt = frag % nfrag;
  int k = kt * 32 + (e & 7) + 8 * (lane >> 4) + 16 * (e >> 3);
  int n = nt * 16 + (lane & 15);
  float v = (n < N) ? B[(size_t)k * N + n] : 0.0f;
  Bp[g] = (_Float16)v;
}

// A fragment: row-major f16 A, lane supplies row m = (lane&15); the two 16B
// chunks cover K offsets {0..7,16..23} (+8 for lanes 16..31).
__device__ __forceinline__ v16h load_afrag(const _Float16* __restrict__ arow,
                                           int k0, int kh) {
  const _Float16* ap = arow + k0 + 8 * kh;
  v8h lo = *(const v8h*)(ap);
  v8h hi = *(const v8h*)(ap + 16);
  v16h a;
#pragma unroll
  for (int i = 0; i < 8; ++i) { a[i] = lo[i]; a[i + 8] = hi[i]; }
  return a;
}

// ---------------------------------------------------------------------------
// WMMA GEMM: C[M][Nreal] = A[M][K](f16,row-major) x Bp(packed f16), fp32 acc.
// Block = 8 waves; wave -> 32x64 strip (8 accumulators); block tile 256x64.
// K is a compile-time multiple of 64: the two-stage ping-pong pipeline is
// fully unrolled, so fragment hand-off is pure SSA renaming (no v_mov chains,
// no WMMA->VALU hazard nops) while next-step loads still issue ahead of the
// current step's WMMAs.
// FUSE=1: C += bias[col] + resid[row][col] (final projection epilogue).
// ---------------------------------------------------------------------------
template <int FUSE, int K>
__global__ __launch_bounds__(256) void wmma_gemm_kernel(
    const _Float16* __restrict__ A, const _Float16* __restrict__ Bp,
    float* __restrict__ C, const float* __restrict__ bias,
    const float* __restrict__ resid, int Npack, int Nreal) {
  const int lane = threadIdx.x & 31;
  const int wave = threadIdx.x >> 5;
  const int m0 = blockIdx.y * 256 + wave * 32;
  const int n0 = blockIdx.x * 64;
  const int idx = lane & 15;
  const int kh  = lane >> 4;

  v8f acc00 = {}, acc01 = {}, acc02 = {}, acc03 = {};
  v8f acc10 = {}, acc11 = {}, acc12 = {}, acc13 = {};
  const _Float16* __restrict__ arow0 = A + (size_t)(m0 + idx) * K;
  const _Float16* __restrict__ arow1 = A + (size_t)(m0 + 16 + idx) * K;
  const _Float16* __restrict__ bcol  = Bp + (size_t)(n0 >> 4) * 512 + lane * 16;
  const size_t bstep = (size_t)(Npack >> 4) * 512;  // halves per k-step of B

#define LOADB(DST0, DST1, DST2, DST3, KK)                                    \
  {                                                                          \
    const _Float16* bb_ = bcol + (size_t)((KK) >> 5) * bstep;                \
    DST0 = *(const v16h*)(bb_);                                              \
    DST1 = *(const v16h*)(bb_ + 512);                                        \
    DST2 = *(const v16h*)(bb_ + 1024);                                       \
    DST3 = *(const v16h*)(bb_ + 1536);                                       \
  }
#define WMMA8(A0, A1, B0, B1, B2, B3)                                        \
  acc00 = __builtin_amdgcn_wmma_f32_16x16x32_f16(false, A0, false, B0, (short)0, acc00, false, false); \
  acc01 = __builtin_amdgcn_wmma_f32_16x16x32_f16(false, A0, false, B1, (short)0, acc01, false, false); \
  acc02 = __builtin_amdgcn_wmma_f32_16x16x32_f16(false, A0, false, B2, (short)0, acc02, false, false); \
  acc03 = __builtin_amdgcn_wmma_f32_16x16x32_f16(false, A0, false, B3, (short)0, acc03, false, false); \
  acc10 = __builtin_amdgcn_wmma_f32_16x16x32_f16(false, A1, false, B0, (short)0, acc10, false, false); \
  acc11 = __builtin_amdgcn_wmma_f32_16x16x32_f16(false, A1, false, B1, (short)0, acc11, false, false); \
  acc12 = __builtin_amdgcn_wmma_f32_16x16x32_f16(false, A1, false, B2, (short)0, acc12, false, false); \
  acc13 = __builtin_amdgcn_wmma_f32_16x16x32_f16(false, A1, false, B3, (short)0, acc13, false, false);

  // stage 0 fragments (k = 0)
  v16h a0 = load_afrag(arow0, 0, kh);
  v16h a1 = load_afrag(arow1, 0, kh);
  v16h b0, b1, b2, b3;
  LOADB(b0, b1, b2, b3, 0)

#pragma unroll
  for (int k0 = 0; k0 < K; k0 += 64) {
    // stage 1 fragments (k0+32) issue before stage-0 WMMAs
    v16h c0 = load_afrag(arow0, k0 + 32, kh);
    v16h c1 = load_afrag(arow1, k0 + 32, kh);
    v16h d0, d1, d2, d3;
    LOADB(d0, d1, d2, d3, k0 + 32)
    __builtin_prefetch(arow0 + k0 + 64, 0, 1);  // global_prefetch_b8
    __builtin_prefetch(arow1 + k0 + 64, 0, 1);

    WMMA8(a0, a1, b0, b1, b2, b3)

    if (k0 + 64 < K) {  // stage 0 fragments for next unrolled iteration
      a0 = load_afrag(arow0, k0 + 64, kh);
      a1 = load_afrag(arow1, k0 + 64, kh);
      LOADB(b0, b1, b2, b3, k0 + 64)
    }
    WMMA8(c0, c1, d0, d1, d2, d3)
  }
#undef WMMA8
#undef LOADB

#define STORE_TILE(ACC, T, MOFF)                                             \
  {                                                                          \
    int col = n0 + (T)*16 + idx;                                             \
    if (col < Nreal) {                                                       \
      _Pragma("unroll") for (int rr = 0; rr < 8; ++rr) {                     \
        int rowi = m0 + (MOFF) + rr + 8 * kh;                                \
        float vv = (ACC)[rr];                                                \
        if (FUSE) vv += bias[col] + resid[(size_t)rowi * Nreal + col];       \
        C[(size_t)rowi * Nreal + col] = vv;                                  \
      }                                                                      \
    }                                                                        \
  }
  STORE_TILE(acc00, 0, 0)  STORE_TILE(acc01, 1, 0)
  STORE_TILE(acc02, 2, 0)  STORE_TILE(acc03, 3, 0)
  STORE_TILE(acc10, 0, 16) STORE_TILE(acc11, 1, 16)
  STORE_TILE(acc12, 2, 16) STORE_TILE(acc13, 3, 16)
#undef STORE_TILE
}

// ---------------------------------------------------------------------------
// Front end: x + pos -> LayerNorm -> {SiLU branch (into cat[:,256:]), RMSNorm
// branch -> u_h (f16)}. One 256-thread block per row, LDS tree reductions.
// ---------------------------------------------------------------------------
__global__ __launch_bounds__(256) void pre_kernel(
    const float* __restrict__ x, const float* __restrict__ pos_w,
    const float* __restrict__ pos_b, const float* __restrict__ lnw,
    const float* __restrict__ lnb, const float* __restrict__ rmsw,
    _Float16* __restrict__ u_h, _Float16* __restrict__ cat_h) {
  __shared__ float red[256];
  int r = blockIdx.x, t = threadIdx.x;
  int l = r & (SEQ - 1);
  float fi = (float)l;
  float denom = (float)(SEQ / 12 + 1);
  float c0  = fi / (float)SEQ;
  float f12 = floorf(fi / 12.0f);
  float c12 = f12 / denom;
  float c34 = (fi - 12.0f * f12) / 12.0f;
  float f56 = floorf((fi + 6.0f) / 12.0f);
  float c56 = f56 / denom;
  float c78 = ((fi + 6.0f) - 12.0f * f56) / 12.0f;
  float pos = c0 * pos_w[t] + c12 * (pos_w[DM + t] + pos_w[2 * DM + t]) +
              c34 * (pos_w[3 * DM + t] + pos_w[4 * DM + t]) +
              c56 * (pos_w[5 * DM + t] + pos_w[6 * DM + t]) +
              c78 * (pos_w[7 * DM + t] + pos_w[8 * DM + t]) + pos_b[t];
  float v = x[(size_t)r * DM + t] + pos;

  red[t] = v; __syncthreads();
  for (int s = 128; s > 0; s >>= 1) { if (t < s) red[t] += red[t + s]; __syncthreads(); }
  float mean = red[0] * (1.0f / DM); __syncthreads();
  float dv = v - mean;
  red[t] = dv * dv; __syncthreads();
  for (int s = 128; s > 0; s >>= 1) { if (t < s) red[t] += red[t + s]; __syncthreads(); }
  float var = red[0] * (1.0f / DM); __syncthreads();
  float xn = dv * rsqrtf(var + EPSF) * lnw[t] + lnb[t];
  cat_h[(size_t)r * (2 * DM) + DM + t] = (_Float16)(xn * sigmf(xn));  // x_silu
  red[t] = xn * xn; __syncthreads();
  for (int s = 128; s > 0; s >>= 1) { if (t < s) red[t] += red[t + s]; __syncthreads(); }
  float ms = red[0] * (1.0f / DM);
  u_h[(size_t)r * DM + t] = (_Float16)(xn * rsqrtf(ms + EPSF) * rmsw[t]);
}

// Causal depthwise conv (k=4) + SiLU on xm = xz[:, :512].
__global__ __launch_bounds__(256) void conv_silu_kernel(
    const float* __restrict__ xz, const float* __restrict__ conv_w,
    const float* __restrict__ conv_b, float* __restrict__ xc,
    _Float16* __restrict__ xc_h) {
  int g = blockIdx.x * 256 + threadIdx.x;
  int d = g & (DI - 1);
  int r = g >> 9;
  int l = r & (SEQ - 1);
  int b0 = r - l;
  float acc = conv_b[d];
#pragma unroll
  for (int k = 0; k < 4; ++k) {
    int ls = l - 3 + k;
    if (ls >= 0) acc += xz[(size_t)(b0 + ls) * 1024 + d] * conv_w[k * DI + d];
  }
  float s = acc * sigmf(acc);
  xc[(size_t)r * DI + d] = s;
  xc_h[(size_t)r * DI + d] = (_Float16)s;
}

// dt = softplus(dbc[:, :16] @ dt_proj_w + dt_proj_b)   (K=16: plain FMA)
__global__ __launch_bounds__(256) void dt_kernel(
    const float* __restrict__ dbc, const float* __restrict__ dtw,
    const float* __restrict__ dtb, float* __restrict__ dt) {
  int g = blockIdx.x * 256 + threadIdx.x;
  int d = g & (DI - 1);
  int r = g >> 9;
  float acc = dtb[d];
#pragma unroll
  for (int j = 0; j < DTR; ++j) acc += dbc[(size_t)r * 48 + j] * dtw[j * DI + d];
  dt[(size_t)r * DI + d] = (acc > 20.0f) ? acc : log1pf(__expf(acc));
}

// Scan pass 1: per-(b,chunk,d) final state from h=0, and sum(dt) over chunk
// (since prod exp(dt*A) == exp(A * sum dt), the chunk decay is one exp).
__global__ __launch_bounds__(256) void scan_chunk_kernel(
    const float* __restrict__ dt, const float* __restrict__ xc,
    const float* __restrict__ dbc, const float* __restrict__ A_log,
    float* __restrict__ Hc, float* __restrict__ Sdt) {
  int g = blockIdx.x * 256 + threadIdx.x;  // b*NCH*DI + c*DI + d
  int d = g & (DI - 1);
  int c = (g >> 9) & (NCH - 1);
  int b = g >> 12;
  float Ad[DS], h[DS];
#pragma unroll
  for (int s = 0; s < DS; ++s) { Ad[s] = -__expf(A_log[d * DS + s]); h[s] = 0.0f; }
  float sdt = 0.0f;
  int row = b * SEQ + c * CHL;
  for (int l = 0; l < CHL; ++l, ++row) {
    float dtv = dt[(size_t)row * DI + d];
    float xv  = xc[(size_t)row * DI + d];
    sdt += dtv;
    float dx = dtv * xv;
    const float* bc = dbc + (size_t)row * 48 + DTR;
#pragma unroll
    for (int s = 0; s < DS; ++s) h[s] = __expf(dtv * Ad[s]) * h[s] + dx * bc[s];
  }
  size_t base = (size_t)g * DS;
#pragma unroll
  for (int s = 0; s < DS; ++s) Hc[base + s] = h[s];
  Sdt[g] = sdt;
}

// Scan pass 2 (tiny serial part): chunk-initial states per (b,d).
__global__ __launch_bounds__(256) void scan_combine_kernel(
    const float* __restrict__ Hc, const float* __restrict__ Sdt,
    const float* __restrict__ A_log, float* __restrict__ Hinit) {
  int g = blockIdx.x * 256 + threadIdx.x;  // b*DI + d
  int d = g & (DI - 1);
  int b = g >> 9;
  float Ad[DS], h[DS];
#pragma unroll
  for (int s = 0; s < DS; ++s) { Ad[s] = -__expf(A_log[d * DS + s]); h[s] = 0.0f; }
  for (int c = 0; c < NCH; ++c) {
    size_t base = (size_t)((b * NCH + c) * DI + d) * DS;
#pragma unroll
    for (int s = 0; s < DS; ++s) Hinit[base + s] = h[s];
    float sdt = Sdt[(b * NCH + c) * DI + d];
#pragma unroll
    for (int s = 0; s < DS; ++s) h[s] = __expf(sdt * Ad[s]) * h[s] + Hc[base + s];
  }
}

// Scan pass 3: outputs with correct initial states; fuse +D*xc, *silu(z), f16.
__global__ __launch_bounds__(256) void scan_output_kernel(
    const float* __restrict__ dt, const float* __restrict__ xc,
    const float* __restrict__ dbc, const float* __restrict__ A_log,
    const float* __restrict__ Hinit, const float* __restrict__ Dp,
    const float* __restrict__ xz, _Float16* __restrict__ y_h) {
  int g = blockIdx.x * 256 + threadIdx.x;
  int d = g & (DI - 1);
  int c = (g >> 9) & (NCH - 1);
  int b = g >> 12;
  float Ad[DS], h[DS];
  size_t base = (size_t)g * DS;
#pragma unroll
  for (int s = 0; s < DS; ++s) { Ad[s] = -__expf(A_log[d * DS + s]); h[s] = Hinit[base + s]; }
  float Dv = Dp[d];
  int row = b * SEQ + c * CHL;
  for (int l = 0; l < CHL; ++l, ++row) {
    float dtv = dt[(size_t)row * DI + d];
    float xv  = xc[(size_t)row * DI + d];
    float dx = dtv * xv;
    const float* bc = dbc + (size_t)row * 48 + DTR;
    const float* cc = dbc + (size_t)row * 48 + DTR + DS;
    float y = 0.0f;
#pragma unroll
    for (int s = 0; s < DS; ++s) {
      h[s] = __expf(dtv * Ad[s]) * h[s] + dx * bc[s];
      y += h[s] * cc[s];
    }
    y += Dv * xv;
    float zv = xz[(size_t)row * 1024 + DI + d];
    y *= zv * sigmf(zv);
    y_h[(size_t)row * DI + d] = (_Float16)y;
  }
}

// LayerNorm(m_out) -> f16 into cat[:, :256].
__global__ __launch_bounds__(256) void lnfwd_kernel(
    const float* __restrict__ m, const float* __restrict__ w,
    const float* __restrict__ bb, _Float16* __restrict__ cat_h) {
  __shared__ float red[256];
  int r = blockIdx.x, t = threadIdx.x;
  float v = m[(size_t)r * DM + t];
  red[t] = v; __syncthreads();
  for (int s = 128; s > 0; s >>= 1) { if (t < s) red[t] += red[t + s]; __syncthreads(); }
  float mean = red[0] * (1.0f / DM); __syncthreads();
  float dv = v - mean;
  red[t] = dv * dv; __syncthreads();
  for (int s = 128; s > 0; s >>= 1) { if (t < s) red[t] += red[t + s]; __syncthreads(); }
  float var = red[0] * (1.0f / DM);
  cat_h[(size_t)r * (2 * DM) + t] = (_Float16)(dv * rsqrtf(var + EPSF) * w[t] + bb[t]);
}

// ---------------------------------------------------------------------------
extern "C" void kernel_launch(void* const* d_in, const int* in_sizes, int n_in,
                              void* d_out, int out_size, void* d_ws, size_t ws_size,
                              hipStream_t stream) {
  const float* x        = (const float*)d_in[0];
  const float* pos_w    = (const float*)d_in[1];
  const float* pos_b    = (const float*)d_in[2];
  const float* ln_in_w  = (const float*)d_in[3];
  const float* ln_in_b  = (const float*)d_in[4];
  const float* rms_w    = (const float*)d_in[5];
  const float* in_proj  = (const float*)d_in[6];
  const float* conv_w   = (const float*)d_in[7];
  const float* conv_b   = (const float*)d_in[8];
  const float* x_proj   = (const float*)d_in[9];
  const float* dt_projw = (const float*)d_in[10];
  const float* dt_projb = (const float*)d_in[11];
  const float* A_log    = (const float*)d_in[12];
  const float* D_param  = (const float*)d_in[13];
  const float* out_proj = (const float*)d_in[14];
  const float* ln_fwd_w = (const float*)d_in[15];
  const float* ln_fwd_b = (const float*)d_in[16];
  const float* proj_w   = (const float*)d_in[17];
  const float* proj_b   = (const float*)d_in[18];
  (void)in_sizes; (void)n_in; (void)out_size; (void)ws_size;

  char* ws = (char*)d_ws;
  size_t off = 0;
  auto alloc = [&](size_t bytes) -> void* {
    void* p = ws + off;
    off = (off + bytes + 255) & ~(size_t)255;
    return p;
  };
  _Float16* u_h   = (_Float16*)alloc((size_t)BL * DM * 2);
  _Float16* cat_h = (_Float16*)alloc((size_t)BL * 2 * DM * 2);
  _Float16* inWp  = (_Float16*)alloc((size_t)DM * 1024 * 2);
  _Float16* xpWp  = (_Float16*)alloc((size_t)DI * 64 * 2);
  _Float16* outWp = (_Float16*)alloc((size_t)DI * DM * 2);
  _Float16* prWp  = (_Float16*)alloc((size_t)(2 * DM) * DM * 2);
  float* xz    = (float*)alloc((size_t)BL * 1024 * 4);
  float* xc    = (float*)alloc((size_t)BL * DI * 4);
  _Float16* xc_h = (_Float16*)alloc((size_t)BL * DI * 2);
  float* dbc   = (float*)alloc((size_t)BL * 48 * 4);
  float* dtbuf = (float*)alloc((size_t)BL * DI * 4);
  float* Hc    = (float*)alloc((size_t)NBATCH * NCH * DI * DS * 4);
  float* Sdt   = (float*)alloc((size_t)NBATCH * NCH * DI * 4);
  float* Hinit = (float*)alloc((size_t)NBATCH * NCH * DI * DS * 4);
  _Float16* y_h = (_Float16*)alloc((size_t)BL * DI * 2);
  float* m_out = (float*)alloc((size_t)BL * DM * 4);

  // Pack all GEMM weights into WMMA-fragment layout (one-shot, tiny).
  auto pack = [&](const float* B, _Float16* Bp, int K, int N, int Np) {
    int total = (K / 32) * (Np / 16) * 512;
    pack_b_kernel<<<(total + 255) / 256, 256, 0, stream>>>(B, Bp, K, N, Np);
  };
  pack(in_proj,  inWp,  DM, 1024, 1024);
  pack(x_proj,   xpWp,  DI, 48,   64);
  pack(out_proj, outWp, DI, DM,   DM);
  pack(proj_w,   prWp,  2 * DM, DM, DM);

  // 1. front end
  pre_kernel<<<BL, 256, 0, stream>>>(x, pos_w, pos_b, ln_in_w, ln_in_b, rms_w, u_h, cat_h);

  // 2. xz = u @ in_proj_w          (32768x256 @ 256x1024)
  wmma_gemm_kernel<0, DM><<<dim3(1024 / 64, BL / 256), 256, 0, stream>>>(
      u_h, inWp, xz, nullptr, nullptr, 1024, 1024);

  // 3. causal conv + silu
  conv_silu_kernel<<<(BL * DI) / 256, 256, 0, stream>>>(xz, conv_w, conv_b, xc, xc_h);

  // 4. dbc = xc @ x_proj_w         (32768x512 @ 512x48, N padded to 64)
  wmma_gemm_kernel<0, DI><<<dim3(1, BL / 256), 256, 0, stream>>>(
      xc_h, xpWp, dbc, nullptr, nullptr, 64, 48);

  // 5. dt = softplus(dbc[:, :16] @ dt_proj_w + b)
  dt_kernel<<<(BL * DI) / 256, 256, 0, stream>>>(dbc, dt_projw, dt_projb, dtbuf);

  // 6. chunked selective scan (3 passes) + gating, f16 output
  scan_chunk_kernel<<<(NBATCH * NCH * DI) / 256, 256, 0, stream>>>(
      dtbuf, xc, dbc, A_log, Hc, Sdt);
  scan_combine_kernel<<<(NBATCH * DI) / 256, 256, 0, stream>>>(Hc, Sdt, A_log, Hinit);
  scan_output_kernel<<<(NBATCH * NCH * DI) / 256, 256, 0, stream>>>(
      dtbuf, xc, dbc, A_log, Hinit, D_param, xz, y_h);

  // 7. m_out = y @ out_proj_w      (32768x512 @ 512x256)
  wmma_gemm_kernel<0, DI><<<dim3(DM / 64, BL / 256), 256, 0, stream>>>(
      y_h, outWp, m_out, nullptr, nullptr, DM, DM);

  // 8. LayerNorm(m_out) -> cat[:, :256]
  lnfwd_kernel<<<BL, 256, 0, stream>>>(m_out, ln_fwd_w, ln_fwd_b, cat_h);

  // 9. out = cat @ proj_w + proj_b + x   (fused epilogue)
  wmma_gemm_kernel<1, 2 * DM><<<dim3(DM / 64, BL / 256), 256, 0, stream>>>(
      cat_h, prWp, (float*)d_out, proj_b, x, DM, DM);
}
